// LSTMCell_21904333209946
// MI455X (gfx1250) — matrix operand order repaired and checked by reading, not compiled, not run on hardware
//
#include <hip/hip_runtime.h>

#define B_  64
#define T_  256
#define D_  32
#define N_  64
#define RS  72   // padded LDS row stride (bf16 elems): 144B, 16B-aligned, conflict-free

typedef __attribute__((ext_vector_type(16))) __bf16 v16bf;
typedef __attribute__((ext_vector_type(8)))  __bf16 v8bf;
typedef __attribute__((ext_vector_type(8)))  float  v8f;

__device__ __forceinline__ float fexp(float x) {
    return __builtin_amdgcn_exp2f(x * 1.44269504088896340736f);
}
__device__ __forceinline__ float sigm(float x) {
    return __builtin_amdgcn_rcpf(1.0f + fexp(-x));
}
__device__ __forceinline__ float tanhx(float x) {
    return 1.0f - 2.0f * __builtin_amdgcn_rcpf(1.0f + fexp(2.0f * x));
}

__global__ __launch_bounds__(128) void lstm_heads_persistent(
    const float* __restrict__ x,
    const float* __restrict__ Wj, const float* __restrict__ Wi,
    const float* __restrict__ Wf, const float* __restrict__ Wo,
    const float* __restrict__ Uj, const float* __restrict__ Ui,
    const float* __restrict__ Uf, const float* __restrict__ Uo,
    const float* __restrict__ bjp, const float* __restrict__ bip,
    const float* __restrict__ bfp, const float* __restrict__ bop,
    float* __restrict__ out)
{
    // h (bf16) double-buffered in LDS: 16 batch rows x 64 cols, padded stride.
    __shared__ __bf16 hlds[2][16 * RS];

    const int blk    = blockIdx.x;        // 0..127
    const int d      = blk & (D_ - 1);    // head
    const int bbase  = (blk >> 5) * 16;   // batch chunk base

    const int tid    = threadIdx.x;
    const int wave   = tid >> 5;          // 0..3 : owns N-cols [16w,16w+16)
    const int lane   = tid & 31;
    const int laneN  = lane & 15;
    const int laneHi = lane >> 4;         // 0/1
    const int col    = wave * 16 + laneN; // output column m

    // ---- Build bf16 B-fragments for 4 gates x 2 K-halves (kept in VGPRs) ----
    // Dense 32x16 B layout: elem e<8 -> K = kh*32 + laneHi*8 + e, e>=8 -> K = kh*32 + 16 + laneHi*8 + (e-8), N = col.
    const float* Wg[4] = {Wj, Wi, Wf, Wo};
    v16bf Bf[4][2];
    #pragma unroll
    for (int g = 0; g < 4; ++g) {
        const float* W = Wg[g] + (size_t)d * N_ * N_ + col;
        #pragma unroll
        for (int kh = 0; kh < 2; ++kh) {
            v16bf bfrag;
            #pragma unroll
            for (int e = 0; e < 16; ++e) {
                int k = kh * 32 + (e < 8 ? laneHi * 8 + e : 16 + laneHi * 8 + (e - 8));     // K (= n) index
                bfrag[e] = (__bf16)W[k * N_];          // W[d, n, m]
            }
            Bf[g][kh] = bfrag;
        }
    }

    // ---- Per-column rank-1 input weights and biases ----
    const int dc = d * N_ + col;
    const float uj = Uj[dc], ui = Ui[dc], uf = Uf[dc], uo = Uo[dc];
    const float cj = bjp[dc], ci = bip[dc], cf = bfp[dc], co = bop[dc];

    // ---- Init state ----
    for (int i = tid; i < 16 * RS; i += 128) {
        hlds[0][i] = (__bf16)0.0f;
        hlds[1][i] = (__bf16)0.0f;
    }
    float cst[8], hst[8];
    #pragma unroll
    for (int r = 0; r < 8; ++r) { cst[r] = 0.0f; hst[r] = 0.0f; }
    __syncthreads();

    int cur = 0;
    #pragma unroll 1
    for (int t = 0; t < T_; ++t) {
        // ---- A fragments (16x32 bf16 each): two contiguous 8-elem runs/lane ----
        // elem e<8  -> K = kh*32 + laneHi*8 + e
        // elem e>=8 -> K = kh*32 + laneHi*8 + 16 + (e-8)
        const __bf16* row = &hlds[cur][laneN * RS];
        v8bf p0 = *(const v8bf*)(row + laneHi * 8);
        v8bf p1 = *(const v8bf*)(row + laneHi * 8 + 16);
        v8bf p2 = *(const v8bf*)(row + laneHi * 8 + 32);
        v8bf p3 = *(const v8bf*)(row + laneHi * 8 + 48);
        v16bf A0 = __builtin_shufflevector(p0, p1, 0,1,2,3,4,5,6,7,8,9,10,11,12,13,14,15);
        v16bf A1 = __builtin_shufflevector(p2, p3, 0,1,2,3,4,5,6,7,8,9,10,11,12,13,14,15);

        // ---- 8x v_wmma_f32_16x16x32_bf16: gate = h @ W (f32 accum) ----
        v8f acc[4];
        #pragma unroll
        for (int g = 0; g < 4; ++g) {
            v8f z = {0.f,0.f,0.f,0.f,0.f,0.f,0.f,0.f};
            z = __builtin_amdgcn_wmma_f32_16x16x32_bf16(false, A0, false, Bf[g][0],
                                                        (short)0, z, false, false);
            z = __builtin_amdgcn_wmma_f32_16x16x32_bf16(false, A1, false, Bf[g][1],
                                                        (short)0, z, false, false);
            acc[g] = z;
        }

        // ---- Elementwise LSTM math; C layout: VGPR r -> M = r + laneHi*8 ----
        const int nxt = cur ^ 1;
        #pragma unroll
        for (int r = 0; r < 8; ++r) {
            const int M = r + laneHi * 8;
            const float xv = x[(size_t)(bbase + M) * (T_ * D_) + t * D_ + d];
            const float gj = acc[0][r] + xv * uj + cj;
            const float gi = acc[1][r] + xv * ui + ci;
            const float gf = acc[2][r] + xv * uf + cf;
            const float go = acc[3][r] + xv * uo + co;
            const float jv = tanhx(gj);
            const float iv = sigm(gi);
            const float fv = sigm(gf);
            const float ov = sigm(go);
            const float cc = cst[r] * fv + iv * jv;
            cst[r] = cc;
            const float hv = ov * tanhx(cc);
            hst[r] = hv;
            out[(size_t)(bbase + M) * (T_ * D_ * N_) + (size_t)t * (D_ * N_)
                + d * N_ + col] = hv;
            hlds[nxt][M * RS + col] = (__bf16)hv;
        }
        __syncthreads();
        cur ^= 1;
    }

    // ---- Final (h_T, c_T) ----
    const size_t OUT0 = (size_t)B_ * T_ * D_ * N_;
    const size_t HN   = (size_t)B_ * D_ * N_;
    #pragma unroll
    for (int r = 0; r < 8; ++r) {
        const int M = r + laneHi * 8;
        const size_t idx = (size_t)(bbase + M) * (D_ * N_) + d * N_ + col;
        out[OUT0 + idx]      = hst[r];
        out[OUT0 + HN + idx] = cst[r];
    }
}

extern "C" void kernel_launch(void* const* d_in, const int* in_sizes, int n_in,
                              void* d_out, int out_size, void* d_ws, size_t ws_size,
                              hipStream_t stream) {
    (void)in_sizes; (void)n_in; (void)out_size; (void)d_ws; (void)ws_size;
    const float* x  = (const float*)d_in[0];
    const float* Wj = (const float*)d_in[1];
    const float* Wi = (const float*)d_in[2];
    const float* Wf = (const float*)d_in[3];
    const float* Wo = (const float*)d_in[4];
    const float* Uj = (const float*)d_in[5];
    const float* Ui = (const float*)d_in[6];
    const float* Uf = (const float*)d_in[7];
    const float* Uo = (const float*)d_in[8];
    const float* bj = (const float*)d_in[9];
    const float* bi = (const float*)d_in[10];
    const float* bf = (const float*)d_in[11];
    const float* bo = (const float*)d_in[12];
    float* out = (float*)d_out;

    dim3 grid(D_ * (B_ / 16));  // 128 independent recurrent streams
    dim3 block(128);            // 4 waves: one 16-col slice each, all 4 gates
    lstm_heads_persistent<<<grid, block, 0, stream>>>(
        x, Wj, Wi, Wf, Wo, Uj, Ui, Uf, Uo, bj, bi, bf, bo, out);
}
